// TripletLoss_41497974014034
// MI455X (gfx1250) — compile-verified, matrix operand order checked by reading, not run on hardware
//
#include <hip/hip_runtime.h>
#include <hip/hip_bf16.h>

typedef __attribute__((ext_vector_type(16))) _Float16 v16h;
typedef __attribute__((ext_vector_type(8)))  _Float16 v8h;
typedef __attribute__((ext_vector_type(8)))  float    v8f;

#define EMB_N 8192
#define EMB_D 512
#define TRIP_T 262144
#define MARGIN_F 1.0f

// ---------------------------------------------------------------------------
// Kernel 0: zero the scalar output (atomic accumulation target)
// ---------------------------------------------------------------------------
__global__ void TL_zero_out(float* out) { *out = 0.0f; }

// ---------------------------------------------------------------------------
// Kernel 1: row-normalize embeddings, store as f16 (8 MB -> L2 resident).
// One wave32 per row; each lane owns 16 contiguous floats of the 512-wide row.
// ---------------------------------------------------------------------------
__global__ __launch_bounds__(256) void TL_normalize_f16(
    const float* __restrict__ emb, _Float16* __restrict__ out) {
  const int wave = threadIdx.x >> 5;
  const int lane = threadIdx.x & 31;
  const int row  = blockIdx.x * 8 + wave;

  const float* p = emb + (size_t)row * EMB_D + lane * 16;
  float x[16];
#pragma unroll
  for (int i = 0; i < 4; ++i) {
    float4 v = ((const float4*)p)[i];
    x[4 * i + 0] = v.x; x[4 * i + 1] = v.y;
    x[4 * i + 2] = v.z; x[4 * i + 3] = v.w;
  }
  float s = 0.0f;
#pragma unroll
  for (int i = 0; i < 16; ++i) s += x[i] * x[i];
  // wave32 butterfly all-reduce
#pragma unroll
  for (int m = 16; m >= 1; m >>= 1) s += __shfl_xor(s, m, 32);
  const float r = 1.0f / sqrtf(s);

  v8h o0, o1;
#pragma unroll
  for (int i = 0; i < 8; ++i) {
    o0[i] = (_Float16)(x[i] * r);
    o1[i] = (_Float16)(x[i + 8] * r);
  }
  _Float16* q = out + (size_t)row * EMB_D + lane * 16;
  *(v8h*)(q)     = o0;
  *(v8h*)(q + 8) = o1;
}

// ---------------------------------------------------------------------------
// Kernel 2: per-wave tile of 16 triplets. Gather 16 anchor / pos / neg rows
// (f16, L2-resident), run two WMMA chains over K=512:
//   Cp = A·Pos^T, Cn = A·Neg^T  (only diagonals are needed)
// Gram-product trick: B operand lane n holds pos/neg ROW n with the same
// K-packing as the A operand -> no transpose loads required.
// ---------------------------------------------------------------------------
__global__ __launch_bounds__(256) void TL_triplet_wmma(
    const _Float16* __restrict__ embf,
    const int* __restrict__ ind_a,
    const int* __restrict__ ind_p,
    const int* __restrict__ ind_n,
    float* __restrict__ out) {
  __shared__ float wsum[8];

  const int wave = threadIdx.x >> 5;
  const int lane = threadIdx.x & 31;
  const int tile = blockIdx.x * 8 + wave;   // 16 triplets per tile
  const int base = tile * 16;
  const int m    = lane & 15;               // both lane halves mirror rows 0..15

  const long ia = ind_a[base + m];
  const long ip = ind_p[base + m];
  const long in_ = ind_n[base + m];
  const _Float16* ra = embf + ia * (long)EMB_D;
  const _Float16* rp = embf + ip * (long)EMB_D;
  const _Float16* rn = embf + in_ * (long)EMB_D;

  // 16-bit A/B fragment K-packing (ISA 7.12.2): lanes 0-15 carry K {0..7,16..23},
  // lanes 16-31 carry K {8..15,24..31} within each 32-wide K step.
  const int off = (lane < 16) ? 0 : 8;

  v8f cp = {};
  v8f cn = {};
#pragma unroll 4
  for (int k = 0; k < EMB_D; k += 32) {
    v8h a0 = *(const v8h*)(ra + k + off);
    v8h a1 = *(const v8h*)(ra + k + 16 + off);
    v8h p0 = *(const v8h*)(rp + k + off);
    v8h p1 = *(const v8h*)(rp + k + 16 + off);
    v8h n0 = *(const v8h*)(rn + k + off);
    v8h n1 = *(const v8h*)(rn + k + 16 + off);
    v16h a  = __builtin_shufflevector(a0, a1, 0,1,2,3,4,5,6,7,8,9,10,11,12,13,14,15);
    v16h bp = __builtin_shufflevector(p0, p1, 0,1,2,3,4,5,6,7,8,9,10,11,12,13,14,15);
    v16h bn = __builtin_shufflevector(n0, n1, 0,1,2,3,4,5,6,7,8,9,10,11,12,13,14,15);
    cp = __builtin_amdgcn_wmma_f32_16x16x32_f16(false, a, false, bp, (short)0, cp, false, false);
    cn = __builtin_amdgcn_wmma_f32_16x16x32_f16(false, a, false, bn, (short)0, cn, false, false);
  }

  // Diagonal of 16x16 f32 C: element i at VGPR (i&7), lane i (i<8) or lane i+16 (i>=8).
  const int r = lane & 7;
  float dp = cp[0];
  float dn = cn[0];
#pragma unroll
  for (int j = 1; j < 8; ++j) {
    dp = (r == j) ? cp[j] : dp;
    dn = (r == j) ? cn[j] : dn;
  }

  const bool relevant = (lane < 8) || (lane >= 24);
  // dist_pos - dist_neg + margin = (1-dp) - (1-dn) + 1 = dn - dp + 1
  float v = relevant ? fmaxf(dn - dp + MARGIN_F, 0.0f) * (1.0f / (float)TRIP_T)
                     : 0.0f;
#pragma unroll
  for (int s = 16; s >= 1; s >>= 1) v += __shfl_xor(v, s, 32);

  if (lane == 0) wsum[wave] = v;
  __syncthreads();
  if (threadIdx.x == 0) {
    float s = 0.0f;
#pragma unroll
    for (int w = 0; w < 8; ++w) s += wsum[w];
    atomicAdd(out, s);
  }
}

// ---------------------------------------------------------------------------
extern "C" void kernel_launch(void* const* d_in, const int* in_sizes, int n_in,
                              void* d_out, int out_size, void* d_ws, size_t ws_size,
                              hipStream_t stream) {
  const float* emb  = (const float*)d_in[0];
  const int*   ia   = (const int*)d_in[1];
  const int*   ip   = (const int*)d_in[2];
  const int*   ineg = (const int*)d_in[3];
  float* out = (float*)d_out;
  _Float16* embf = (_Float16*)d_ws;   // 8192*512*2 = 8 MB scratch

  TL_zero_out<<<1, 1, 0, stream>>>(out);
  TL_normalize_f16<<<EMB_N / 8, 256, 0, stream>>>(emb, embf);
  TL_triplet_wmma<<<TRIP_T / 16 / 8, 256, 0, stream>>>(embf, ia, ip, ineg, out);
}
